// SinkhornDistance_26388279067457
// MI455X (gfx1250) — matrix operand order repaired and checked by reading, not compiled, not run on hardware
//
#include <hip/hip_runtime.h>
#include <hip/hip_bf16.h>

#define BATCH 16
#define NPTS  1024
#define DIM   1024
#define EPS_SINK 1.0e-3f
#define MAX_ITER 100

typedef __attribute__((ext_vector_type(16))) _Float16 v16h;
typedef __attribute__((ext_vector_type(8)))  _Float16 v8h;
typedef __attribute__((ext_vector_type(4)))  _Float16 v4h;
typedef __attribute__((ext_vector_type(8)))  float    v8f;

// ---------------------------------------------------------------------------
// init: u = v = 1/N, max accumulator = 0, out = 0
// ---------------------------------------------------------------------------
__global__ void sink_init(float* __restrict__ u, float* __restrict__ v,
                          unsigned int* __restrict__ maxbits, float* __restrict__ out) {
    int i = blockIdx.x * blockDim.x + threadIdx.x;
    if (i < BATCH * NPTS) {
        u[i] = 1.0f / (float)NPTS;
        v[i] = 1.0f / (float)NPTS;
    }
    if (i == 0) { *maxbits = 0u; out[0] = 0.0f; }
}

// ---------------------------------------------------------------------------
// normalize rows of x and y (f32 [B,N,D]) -> unit rows stored as f16
// one block (256 threads) per row; D = 1024 = 256 float4
// ---------------------------------------------------------------------------
__global__ void normalize_rows(const float* __restrict__ x, const float* __restrict__ y,
                               _Float16* __restrict__ Xh, _Float16* __restrict__ Yh) {
    const int row = blockIdx.x;
    const int NR  = BATCH * NPTS;
    const float*    src = (row < NR) ? (x + (size_t)row * DIM) : (y + (size_t)(row - NR) * DIM);
    _Float16*       dst = (row < NR) ? (Xh + (size_t)row * DIM) : (Yh + (size_t)(row - NR) * DIM);

    const float4 val = ((const float4*)src)[threadIdx.x];
    float ss = val.x * val.x + val.y * val.y + val.z * val.z + val.w * val.w;

    __shared__ float red[256];
    red[threadIdx.x] = ss;
    __syncthreads();
    #pragma unroll
    for (int off = 128; off > 0; off >>= 1) {
        if (threadIdx.x < off) red[threadIdx.x] += red[threadIdx.x + off];
        __syncthreads();
    }
    const float inv = 1.0f / fmaxf(sqrtf(red[0]), 1e-12f);

    v4h h;
    h.x = (_Float16)(val.x * inv);
    h.y = (_Float16)(val.y * inv);
    h.z = (_Float16)(val.z * inv);
    h.w = (_Float16)(val.w * inv);
    ((v4h*)dst)[threadIdx.x] = h;
}

// ---------------------------------------------------------------------------
// WMMA fragment loaders (layouts per CDNA5 ISA 7.12.2, 16-bit A 16x32 / B 32x16)
// A per lane: row = lane&15 ; halfs {k + hs*8 .. +7} and {k + hs*8 + 16 .. +7}
// B per lane: Y-row = lane&15 ; halfs {k + hs*8 .. +7} and {k + hs*8 + 16 .. +7}
// ---------------------------------------------------------------------------
__device__ __forceinline__ v16h load_frag_a(const _Float16* p) {
    v8h lo = *(const v8h*)(p);
    v8h hi = *(const v8h*)(p + 16);
    return __builtin_shufflevector(lo, hi, 0,1,2,3,4,5,6,7,8,9,10,11,12,13,14,15);
}
__device__ __forceinline__ v16h load_frag_b(const _Float16* p) {
    v8h lo = *(const v8h*)(p);
    v8h hi = *(const v8h*)(p + 16);
    return __builtin_shufflevector(lo, hi, 0,1,2,3,4,5,6,7,8,9,10,11,12,13,14,15);
}

// ---------------------------------------------------------------------------
// cost[b,n,m] = 1 - Xf[b,n,:] . Yf[b,m,:]   via v_wmma_f32_16x16x32_f16
// one wave computes a 32x32 tile (2x2 WMMA tiles); also reduces global max
// grid: 16 batches * (1024/32)^2 tiles = 16384 waves = 2048 blocks * 8 waves
// ---------------------------------------------------------------------------
__global__ void cost_gemm_wmma(const _Float16* __restrict__ Xh, const _Float16* __restrict__ Yh,
                               float* __restrict__ C, unsigned int* __restrict__ maxbits) {
    const int wave = (blockIdx.x * blockDim.x + threadIdx.x) >> 5;
    const int lane = threadIdx.x & 31;
    const int b  = wave >> 10;
    const int t  = wave & 1023;
    const int n0 = (t >> 5) << 5;   // tile row origin
    const int m0 = (t & 31) << 5;   // tile col origin
    const int r  = lane & 15;
    const int hs = lane >> 4;       // half-wave selector

    const _Float16* a0 = Xh + ((size_t)(b * NPTS + n0 + r) * DIM) + hs * 8;
    const _Float16* a1 = a0 + (size_t)16 * DIM;
    const _Float16* b0 = Yh + ((size_t)(b * NPTS + m0 + r) * DIM) + hs * 8;
    const _Float16* b1 = b0 + (size_t)16 * DIM;

    v8f acc00 = {}; v8f acc01 = {}; v8f acc10 = {}; v8f acc11 = {};

    for (int k = 0; k < DIM; k += 32) {
        v16h A0 = load_frag_a(a0 + k);
        v16h A1 = load_frag_a(a1 + k);
        v16h B0 = load_frag_b(b0 + k);
        v16h B1 = load_frag_b(b1 + k);
        acc00 = __builtin_amdgcn_wmma_f32_16x16x32_f16(false, A0, false, B0, (short)0, acc00, false, false);
        acc01 = __builtin_amdgcn_wmma_f32_16x16x32_f16(false, A0, false, B1, (short)0, acc01, false, false);
        acc10 = __builtin_amdgcn_wmma_f32_16x16x32_f16(false, A1, false, B0, (short)0, acc10, false, false);
        acc11 = __builtin_amdgcn_wmma_f32_16x16x32_f16(false, A1, false, B1, (short)0, acc11, false, false);
    }

    // epilogue: cost = 1 - dot ; D-layout: VGPR j -> row j + hs*8, col = lane&15
    float lmax = 0.0f;
    #pragma unroll
    for (int j = 0; j < 8; ++j) {
        const int rr = hs * 8 + j;
        float c00 = 1.0f - acc00[j];
        float c01 = 1.0f - acc01[j];
        float c10 = 1.0f - acc10[j];
        float c11 = 1.0f - acc11[j];
        C[((size_t)(b * NPTS + n0 + rr)      * NPTS) + (m0 + r)]      = c00;
        C[((size_t)(b * NPTS + n0 + rr)      * NPTS) + (m0 + 16 + r)] = c01;
        C[((size_t)(b * NPTS + n0 + 16 + rr) * NPTS) + (m0 + r)]      = c10;
        C[((size_t)(b * NPTS + n0 + 16 + rr) * NPTS) + (m0 + 16 + r)] = c11;
        lmax = fmaxf(lmax, fmaxf(fmaxf(c00, c01), fmaxf(c10, c11)));
    }
    #pragma unroll
    for (int off = 16; off > 0; off >>= 1)
        lmax = fmaxf(lmax, __shfl_xor(lmax, off, 32));
    if (lane == 0)
        atomicMax(maxbits, __float_as_uint(lmax));   // cost > 0 always -> bit order == float order
}

// ---------------------------------------------------------------------------
// u[b,n] = 1 / (s * sum_m C[b,n,m] v[b,m] + eps) ;  s = 1/maxcost
// one wave per row; float4 coalesced loads; shuffle reduction
// ---------------------------------------------------------------------------
__global__ void u_update(const float* __restrict__ C, const float* __restrict__ v,
                         float* __restrict__ u, const unsigned int* __restrict__ maxbits) {
    const int wave = (blockIdx.x * blockDim.x + threadIdx.x) >> 5;   // = b*N + n
    const int lane = threadIdx.x & 31;
    const int b = wave >> 10;

    const float4* row = (const float4*)(C + (size_t)wave * NPTS);
    const float4* vp  = (const float4*)(v + (size_t)b * NPTS);
    float acc = 0.0f;
    #pragma unroll
    for (int i = 0; i < 8; ++i) {
        float4 c  = row[lane + i * 32];
        float4 vv = vp[lane + i * 32];
        acc += c.x * vv.x + c.y * vv.y + c.z * vv.z + c.w * vv.w;
    }
    #pragma unroll
    for (int off = 16; off > 0; off >>= 1)
        acc += __shfl_xor(acc, off, 32);
    if (lane == 0) {
        const float s = 1.0f / __uint_as_float(*maxbits);
        u[wave] = 1.0f / (s * acc + EPS_SINK);
    }
}

// ---------------------------------------------------------------------------
// v[b,m] = 1 / (s * sum_n C[b,n,m] u[b,n] + eps)
// one thread per column (coalesced across lanes); u broadcast is uniform
// grid: 16 batches * 4 chunks of 256 columns = 64 blocks
// ---------------------------------------------------------------------------
__global__ void v_update(const float* __restrict__ C, const float* __restrict__ u,
                         float* __restrict__ v, const unsigned int* __restrict__ maxbits) {
    const int b = blockIdx.x >> 2;
    const int m = ((blockIdx.x & 3) << 8) + threadIdx.x;
    const float* col = C + (size_t)b * NPTS * NPTS + m;
    const float* up  = u + (size_t)b * NPTS;

    float acc = 0.0f;
    #pragma unroll 8
    for (int n = 0; n < NPTS; ++n)
        acc += col[(size_t)n * NPTS] * up[n];

    const float s = 1.0f / __uint_as_float(*maxbits);
    v[(size_t)b * NPTS + m] = 1.0f / (s * acc + EPS_SINK);
}

// ---------------------------------------------------------------------------
// w[b*N+n] = u[b,n] * s * sum_m C[b,n,m] v[b,m]   (per-row contribution)
// ---------------------------------------------------------------------------
__global__ void row_contrib(const float* __restrict__ C, const float* __restrict__ v,
                            const float* __restrict__ u, const unsigned int* __restrict__ maxbits,
                            float* __restrict__ w) {
    const int wave = (blockIdx.x * blockDim.x + threadIdx.x) >> 5;
    const int lane = threadIdx.x & 31;
    const int b = wave >> 10;

    const float4* row = (const float4*)(C + (size_t)wave * NPTS);
    const float4* vp  = (const float4*)(v + (size_t)b * NPTS);
    float acc = 0.0f;
    #pragma unroll
    for (int i = 0; i < 8; ++i) {
        float4 c  = row[lane + i * 32];
        float4 vv = vp[lane + i * 32];
        acc += c.x * vv.x + c.y * vv.y + c.z * vv.z + c.w * vv.w;
    }
    #pragma unroll
    for (int off = 16; off > 0; off >>= 1)
        acc += __shfl_xor(acc, off, 32);
    if (lane == 0) {
        const float s = 1.0f / __uint_as_float(*maxbits);
        w[wave] = u[wave] * s * acc;
    }
}

// ---------------------------------------------------------------------------
// deterministic fixed-order final reduction: out = sum(w) / BATCH
// ---------------------------------------------------------------------------
__global__ void final_reduce(const float* __restrict__ w, float* __restrict__ out) {
    __shared__ float red[256];
    float s = 0.0f;
    for (int i = threadIdx.x; i < BATCH * NPTS; i += 256) s += w[i];
    red[threadIdx.x] = s;
    __syncthreads();
    #pragma unroll
    for (int off = 128; off > 0; off >>= 1) {
        if (threadIdx.x < off) red[threadIdx.x] += red[threadIdx.x + off];
        __syncthreads();
    }
    if (threadIdx.x == 0) out[0] = red[0] * (1.0f / (float)BATCH);
}

// ---------------------------------------------------------------------------
extern "C" void kernel_launch(void* const* d_in, const int* in_sizes, int n_in,
                              void* d_out, int out_size, void* d_ws, size_t ws_size,
                              hipStream_t stream) {
    (void)in_sizes; (void)n_in; (void)out_size; (void)ws_size;

    const float* x = (const float*)d_in[0];
    const float* y = (const float*)d_in[1];
    float* out = (float*)d_out;

    // workspace layout
    char* ws = (char*)d_ws;
    float*        C    = (float*)ws;                                  // 64 MB
    _Float16*     Xh   = (_Float16*)(ws + (size_t)BATCH * NPTS * NPTS * sizeof(float));
    _Float16*     Yh   = Xh + (size_t)BATCH * NPTS * DIM;             // 32 MB each
    float*        u    = (float*)(Yh + (size_t)BATCH * NPTS * DIM);
    float*        v    = u + BATCH * NPTS;
    float*        w    = v + BATCH * NPTS;
    unsigned int* maxb = (unsigned int*)(w + BATCH * NPTS);

    sink_init<<<dim3(64), dim3(256), 0, stream>>>(u, v, maxb, out);
    normalize_rows<<<dim3(2 * BATCH * NPTS), dim3(256), 0, stream>>>(x, y, Xh, Yh);
    cost_gemm_wmma<<<dim3(2048), dim3(256), 0, stream>>>(Xh, Yh, C, maxb);

    for (int it = 0; it < MAX_ITER; ++it) {
        u_update<<<dim3(2048), dim3(256), 0, stream>>>(C, v, u, maxb);
        v_update<<<dim3(64),   dim3(256), 0, stream>>>(C, u, v, maxb);
    }

    row_contrib<<<dim3(2048), dim3(256), 0, stream>>>(C, v, u, maxb, w);
    final_reduce<<<dim3(1), dim3(256), 0, stream>>>(w, out);
}
